// TGAT_90632399880282
// MI455X (gfx1250) — matrix operand order repaired and checked by reading, not compiled, not run on hardware
//
#include <hip/hip_runtime.h>
#include <hip/hip_bf16.h>
#include <math.h>

#define N_NODES 10000
#define E_EDGES 160000
#define EFULL   (E_EDGES + N_NODES)
#define SEQ     32
#define IN_CH   128
#define HID     64
#define HEADS   8
#define HC1     (HEADS * HID) /* 512 */
#define GRU_H   256
#define NEG_SLOPE 0.2f

typedef __attribute__((ext_vector_type(16))) _Float16 v16h;
typedef __attribute__((ext_vector_type(8)))  _Float16 h8;
typedef __attribute__((ext_vector_type(8)))  float    v8f;
typedef __attribute__((ext_vector_type(4)))  float    f32x4;

// ---------------------------------------------------------------------------
// order-preserving float<->uint map so segment-max can use u32 atomicMax
// ---------------------------------------------------------------------------
__device__ __forceinline__ unsigned int enc_ord(float f) {
  unsigned int u = __float_as_uint(f);
  return (u & 0x80000000u) ? ~u : (u | 0x80000000u);
}
__device__ __forceinline__ float dec_ord(unsigned int u) {
  return __uint_as_float((u & 0x80000000u) ? (u & 0x7FFFFFFFu) : ~u);
}

// ---------------------------------------------------------------------------
// utility zero kernels
// ---------------------------------------------------------------------------
__global__ void zero_f32(float* p, int n) {
  int i = blockIdx.x * blockDim.x + threadIdx.x;
  if (i < n) p[i] = 0.0f;
}
__global__ void zero_u32(unsigned int* p, int n) {
  int i = blockIdx.x * blockDim.x + threadIdx.x;
  if (i < n) p[i] = 0u;
}

// ---------------------------------------------------------------------------
// WMMA GEMM:  C[M,N] = A[M,K] * B(.T) + bias   (fp32 in/out, f16 matrix core)
// PRECONDITIONS (all call sites satisfy them): K % 32 == 0, N % 64 == 0,
// A/B rows 16B-aligned.  Only M may be ragged.
// tile: 128(M) x 64(N), K stepped by 32.  8 waves/block, wave = 16 x 64.
// LDS tiles stored FRAGMENT-MAJOR so each WMMA fragment is two ds_load_b128:
//  - A row k-groups permuted [g0,g2,g1,g3]: lane half h reads 32B at h*32.
//  - B stored transposed: row n holds k=0..31 contiguously.
// Rows padded to 40 halves (80B = 5*16B) to keep 16B alignment.
// ---------------------------------------------------------------------------
template <int TRANSB>
__global__ __launch_bounds__(256) void gemm_wmma_f16(
    const float* __restrict__ A, const float* __restrict__ B,
    const float* __restrict__ bias, float* __restrict__ C,
    int M, int N, int K)
{
  __shared__ _Float16 Ash[128][40];   // [m][permuted k halves 0..31], pad to 40
  __shared__ _Float16 Bsh[64][40];    // [n][k halves 0..31], pad to 40

  const int tid  = threadIdx.x;
  const int lane = tid & 31;
  const int wave = tid >> 5;          // 0..7
  const int half = lane >> 4;         // 0/1
  const int lmod = lane & 15;
  const int m0   = blockIdx.y * 128;
  const int n0   = blockIdx.x * 64;

  v8f acc[4] = {};

  for (int k0 = 0; k0 < K; k0 += 32) {
    // ---- stage A tile (128 rows x 32 k): 4 x b128 global, 2 x b128 LDS ----
    {
      int row  = tid >> 1;
      int kb16 = tid & 1;             // which 16-k chunk of this row
      int gm   = m0 + row;
      h8 lo, hi;
      if (gm < M) {                   // uniform per-thread branch
        const f32x4* ap = (const f32x4*)(A + (size_t)gm * K + k0 + kb16 * 16);
        f32x4 v0 = ap[0], v1 = ap[1], v2 = ap[2], v3 = ap[3];
        #pragma unroll
        for (int i = 0; i < 4; ++i) {
          lo[i]     = (_Float16)v0[i];
          lo[i + 4] = (_Float16)v1[i];
          hi[i]     = (_Float16)v2[i];
          hi[i + 4] = (_Float16)v3[i];
        }
        if (k0 + 32 < K)
          __builtin_prefetch(A + (size_t)gm * K + k0 + 32 + kb16 * 16, 0, 3);
      } else {
        #pragma unroll
        for (int i = 0; i < 8; ++i) { lo[i] = (_Float16)0.0f; hi[i] = (_Float16)0.0f; }
      }
      // k-group -> LDS slot permutation {0,2,1,3}
      int s0 = kb16 ? 1 : 0;
      int s1 = kb16 ? 3 : 2;
      *(h8*)&Ash[row][s0 * 8] = lo;
      *(h8*)&Ash[row][s1 * 8] = hi;
    }
    // ---- stage B tile transposed into Bsh[n][k] ----
    if (TRANSB) {
      // B is [N][K]: thread reads 8 consecutive k of one n -> one b128 store
      int nloc = tid >> 2;            // 0..63
      int kb8  = tid & 3;             // which 8-k group
      int gn   = n0 + nloc;           // always < N (N % 64 == 0)
      const f32x4* bp = (const f32x4*)(B + (size_t)gn * K + k0 + kb8 * 8);
      f32x4 v0 = bp[0], v1 = bp[1];
      h8 bv;
      #pragma unroll
      for (int i = 0; i < 4; ++i) {
        bv[i]     = (_Float16)v0[i];
        bv[i + 4] = (_Float16)v1[i];
      }
      *(h8*)&Bsh[nloc][kb8 * 8] = bv;
    } else {
      // B is [K][N]: coalesced wide read along n, scattered b16 stores
      int krow = tid >> 3;            // 0..31, gk always < K (K % 32 == 0)
      int nb   = (tid & 7) * 8;
      int gk   = k0 + krow;
      const f32x4* bp = (const f32x4*)(B + (size_t)gk * N + n0 + nb);
      f32x4 v0 = bp[0], v1 = bp[1];
      #pragma unroll
      for (int i = 0; i < 4; ++i) {
        Bsh[nb + i][krow]     = (_Float16)v0[i];
        Bsh[nb + 4 + i][krow] = (_Float16)v1[i];
      }
    }
    __syncthreads();

    // ---- A fragment: 2 x b128 + register shuffle ----
    int mrow = wave * 16 + lmod;
    h8 a0 = *(const h8*)&Ash[mrow][half * 16];
    h8 a1 = *(const h8*)&Ash[mrow][half * 16 + 8];
    v16h afrag = __builtin_shufflevector(a0, a1,
        0, 1, 2, 3, 4, 5, 6, 7, 8, 9, 10, 11, 12, 13, 14, 15);

    // ---- preload all 4 B fragments, then 4 back-to-back WMMAs ----
    h8 bl[4], bh[4];
    #pragma unroll
    for (int t = 0; t < 4; ++t) {
      const _Float16* bp = &Bsh[t * 16 + lmod][half * 8];
      bl[t] = *(const h8*)bp;
      bh[t] = *(const h8*)(bp + 16);
    }
    #pragma unroll
    for (int t = 0; t < 4; ++t) {
      v16h bfrag = __builtin_shufflevector(bl[t], bh[t],
          0, 1, 2, 3, 4, 5, 6, 7, 8, 9, 10, 11, 12, 13, 14, 15);
      acc[t] = __builtin_amdgcn_wmma_f32_16x16x32_f16(
          false, afrag, false, bfrag, (short)0, acc[t], false, false);
    }
    __syncthreads();
  }

  // ---- store 16x64 per wave (C/D layout: VGPR r -> M = r + 8*half) ----
  #pragma unroll
  for (int t = 0; t < 4; ++t) {
    #pragma unroll
    for (int r = 0; r < 8; ++r) {
      int m = m0 + wave * 16 + half * 8 + r;
      int n = n0 + t * 16 + lmod;
      if (m < M) {
        float v = acc[t][r];
        if (bias) v += bias[n];
        C[(size_t)m * N + n] = v;
      }
    }
  }
}

// ---------------------------------------------------------------------------
// graph setup: degree, mean self-loop attr, full edge arrays
// ---------------------------------------------------------------------------
__global__ void setup_edges(const float* __restrict__ edge_attr,
                            const int* __restrict__ esrc, const int* __restrict__ edst,
                            int* deg, float* loopsum,
                            int* src_full, int* dst_full, float* ea_full, int E)
{
  int e = blockIdx.x * blockDim.x + threadIdx.x;
  if (e >= E) return;
  int s = esrc[e], d = edst[e];
  src_full[e] = s;
  dst_full[e] = d;
  atomicAdd(&deg[d], 1);
  #pragma unroll
  for (int a = 0; a < 3; ++a) {
    float v = edge_attr[e * 3 + a];
    ea_full[e * 3 + a] = v;
    atomicAdd(&loopsum[d * 3 + a], v);
  }
}

__global__ void setup_loops(int* src_full, int* dst_full, float* ea_full,
                            const int* __restrict__ deg,
                            const float* __restrict__ loopsum, int E, int N)
{
  int n = blockIdx.x * blockDim.x + threadIdx.x;
  if (n >= N) return;
  src_full[E + n] = n;
  dst_full[E + n] = n;
  float dv = deg[n] > 0 ? (float)deg[n] : 1.0f;
  #pragma unroll
  for (int a = 0; a < 3; ++a)
    ea_full[(size_t)(E + n) * 3 + a] = loopsum[n * 3 + a] / dv;
}

// single-block scan: rowptr over (deg[n] + 1) incoming edges per node
__global__ void scan_rowptr(const int* __restrict__ deg, int* rowptr, int N)
{
  __shared__ int sums[256];
  const int tid = threadIdx.x;
  const int nth = 256;
  int chunk = (N + nth - 1) / nth;
  int b = tid * chunk; if (b > N) b = N;
  int e = b + chunk;   if (e > N) e = N;
  int s = 0;
  for (int i = b; i < e; ++i) s += deg[i] + 1;
  sums[tid] = s;
  __syncthreads();
  for (int off = 1; off < nth; off <<= 1) {
    int v = (tid >= off) ? sums[tid - off] : 0;
    __syncthreads();
    sums[tid] += v;
    __syncthreads();
  }
  int run = sums[tid] - s;   // exclusive prefix
  for (int i = b; i < e; ++i) { rowptr[i] = run; run += deg[i] + 1; }
  if (e == N && b < N) rowptr[N] = run;
}

__global__ void fill_csr(const int* __restrict__ dst_full,
                         const int* __restrict__ rowptr,
                         int* fillpos, int* eid, int Ef)
{
  int e = blockIdx.x * blockDim.x + threadIdx.x;
  if (e >= Ef) return;
  int d = dst_full[e];
  int idx = rowptr[d] + atomicAdd(&fillpos[d], 1);
  eid[idx] = e;
}

// ---------------------------------------------------------------------------
// GATv2 edge pass 1: per-edge per-head attention logit + segment max.
// one wave per edge; edge embedding ea@we recomputed on the fly (3 FMAs).
// ---------------------------------------------------------------------------
__global__ void gat_edge_logits(const float* __restrict__ xl, const float* __restrict__ xr,
                                const float* __restrict__ ea,
                                const int* __restrict__ srcf, const int* __restrict__ dstf,
                                const float* __restrict__ we, const float* __restrict__ att,
                                float* __restrict__ logits, unsigned int* __restrict__ mx,
                                int Ef, int H, int C)
{
  int wid = (int)blockIdx.x * (blockDim.x >> 5) + ((int)threadIdx.x >> 5);
  if (wid >= Ef) return;
  int lane = threadIdx.x & 31;
  int s = srcf[wid], d = dstf[wid];
  float a0 = ea[(size_t)wid * 3 + 0];
  float a1 = ea[(size_t)wid * 3 + 1];
  float a2 = ea[(size_t)wid * 3 + 2];
  int HC = H * C;
  const float* xls = xl + (size_t)s * HC;
  const float* xrd = xr + (size_t)d * HC;
  for (int h = 0; h < H; ++h) {
    float p = 0.0f;
    for (int c = lane; c < C; c += 32) {
      int idx = h * C + c;
      float ee = a0 * we[idx] + a1 * we[HC + idx] + a2 * we[2 * HC + idx];
      float v = xls[idx] + xrd[idx] + ee;
      v = v > 0.0f ? v : NEG_SLOPE * v;   // leaky relu
      p += v * att[idx];
    }
    for (int off = 16; off > 0; off >>= 1) p += __shfl_xor(p, off, 32);
    if (lane == 0) {
      logits[(size_t)wid * H + h] = p;
      atomicMax(&mx[(size_t)d * H + h], enc_ord(p));
    }
  }
}

// edge pass 2: exp(logit - max) + segment sum (denominator)
__global__ void gat_edge_softmax(const int* __restrict__ dstf,
                                 const unsigned int* __restrict__ mx,
                                 float* __restrict__ exv, float* __restrict__ den,
                                 int Ef, int H)
{
  int i = blockIdx.x * blockDim.x + threadIdx.x;
  if (i >= Ef * H) return;
  int e = i / H, h = i - e * H;
  int d = dstf[e];
  float ex = __expf(exv[i] - dec_ord(mx[(size_t)d * H + h]));
  exv[i] = ex;
  atomicAdd(&den[(size_t)d * H + h], ex);
}

// edge pass 3: CSR aggregation per node + bias + ELU.  block = node, thread = channel.
__global__ void gat_aggregate(const float* __restrict__ xl, const float* __restrict__ exv,
                              const float* __restrict__ den,
                              const int* __restrict__ rowptr, const int* __restrict__ eid,
                              const int* __restrict__ srcf,
                              const float* __restrict__ bias, float* __restrict__ out,
                              int H, int C)
{
  int n = blockIdx.x;
  int c = threadIdx.x;          // blockDim == H*C
  int HC = H * C;
  int h = c / C;
  float invden = 1.0f / den[(size_t)n * H + h];
  float acc = 0.0f;
  int beg = rowptr[n], end = rowptr[n + 1];
  for (int i = beg; i < end; ++i) {
    int e = eid[i];
    float w = exv[(size_t)e * H + h] * invden;
    acc += w * xl[(size_t)srcf[e] * HC + c];
  }
  acc += bias[c];
  out[(size_t)n * HC + c] = acc > 0.0f ? acc : expm1f(acc);   // ELU
}

// ---------------------------------------------------------------------------
// GRU cell (gi/gh already GEMM'ed with biases)
// ---------------------------------------------------------------------------
__global__ void gru_cell(const float* __restrict__ gi, const float* __restrict__ gh,
                         float* __restrict__ h, int N, int Hh)
{
  int i = blockIdx.x * blockDim.x + threadIdx.x;
  if (i >= N * Hh) return;
  int n = i / Hh, j = i - n * Hh;
  const float* gin = gi + (size_t)n * 3 * Hh;
  const float* ghn = gh + (size_t)n * 3 * Hh;
  float r  = 1.0f / (1.0f + __expf(-(gin[j] + ghn[j])));
  float z  = 1.0f / (1.0f + __expf(-(gin[Hh + j] + ghn[Hh + j])));
  float nn = tanhf(gin[2 * Hh + j] + r * ghn[2 * Hh + j]);
  float hv = h[i];
  h[i] = (1.0f - z) * nn + z * hv;
}

// column means of [hf | hb]  -> g[512]
__global__ void col_mean(const float* __restrict__ hf, const float* __restrict__ hb,
                         float* __restrict__ g, int N, int Hh)
{
  int j = blockIdx.x;               // 0..2*Hh-1
  const float* src = (j < Hh) ? (hf + j) : (hb + (j - Hh));
  float s = 0.0f;
  for (int n = threadIdx.x; n < N; n += blockDim.x) s += src[(size_t)n * Hh];
  __shared__ float red[256];
  red[threadIdx.x] = s;
  __syncthreads();
  for (int off = 128; off > 0; off >>= 1) {
    if ((int)threadIdx.x < off) red[threadIdx.x] += red[threadIdx.x + off];
    __syncthreads();
  }
  if (threadIdx.x == 0) g[j] = red[0] / (float)N;
}

__global__ void fc_out(const float* __restrict__ g, const float* __restrict__ w,
                       const float* __restrict__ b, float* __restrict__ out)
{
  int o = threadIdx.x;
  if (o < 33) {
    float s = b[o];
    for (int k = 0; k < 2 * GRU_H; ++k) s += g[k] * w[o * (2 * GRU_H) + k];
    out[o] = s;
  }
}

// ---------------------------------------------------------------------------
// host launcher
// ---------------------------------------------------------------------------
extern "C" void kernel_launch(void* const* d_in, const int* in_sizes, int n_in,
                              void* d_out, int out_size, void* d_ws, size_t ws_size,
                              hipStream_t stream)
{
  (void)in_sizes; (void)n_in; (void)out_size; (void)ws_size;
  const float* x      = (const float*)d_in[0];
  const float* eattr  = (const float*)d_in[1];
  const int*   esrc   = (const int*)d_in[2];
  const int*   edst   = (const int*)d_in[3];
  const float* g1wl   = (const float*)d_in[4];
  const float* g1wr   = (const float*)d_in[5];
  const float* g1we   = (const float*)d_in[6];
  const float* g1att  = (const float*)d_in[7];
  const float* g1b    = (const float*)d_in[8];
  const float* g2wl   = (const float*)d_in[9];
  const float* g2wr   = (const float*)d_in[10];
  const float* g2we   = (const float*)d_in[11];
  const float* g2att  = (const float*)d_in[12];
  const float* g2b    = (const float*)d_in[13];
  const float* wih_f  = (const float*)d_in[14];
  const float* whh_f  = (const float*)d_in[15];
  const float* bih_f  = (const float*)d_in[16];
  const float* bhh_f  = (const float*)d_in[17];
  const float* wih_b  = (const float*)d_in[18];
  const float* whh_b  = (const float*)d_in[19];
  const float* bih_b  = (const float*)d_in[20];
  const float* bhh_b  = (const float*)d_in[21];
  const float* fcw    = (const float*)d_in[22];
  const float* fcb    = (const float*)d_in[23];
  float* out = (float*)d_out;

  // ---- carve workspace ----
  char* p = (char*)d_ws;
  auto carve = [&](size_t bytes) -> void* {
    void* r = (void*)p;
    p += (bytes + 255) & ~(size_t)255;
    return r;
  };
  int*   src_full = (int*)  carve(sizeof(int) * EFULL);
  int*   dst_full = (int*)  carve(sizeof(int) * EFULL);
  float* ea_full  = (float*)carve(sizeof(float) * (size_t)EFULL * 3);
  int*   deg      = (int*)  carve(sizeof(int) * N_NODES);
  float* loopsum  = (float*)carve(sizeof(float) * N_NODES * 3);
  int*   rowptr   = (int*)  carve(sizeof(int) * (N_NODES + 1));
  int*   fillpos  = (int*)  carve(sizeof(int) * N_NODES);
  int*   eid      = (int*)  carve(sizeof(int) * EFULL);
  float* xl       = (float*)carve(sizeof(float) * (size_t)N_NODES * HC1);
  float* xr       = (float*)carve(sizeof(float) * (size_t)N_NODES * HC1);
  float* h1       = (float*)carve(sizeof(float) * (size_t)N_NODES * HC1);
  unsigned int* mx = (unsigned int*)carve(sizeof(unsigned int) * (size_t)N_NODES * HEADS);
  float* den      = (float*)carve(sizeof(float) * (size_t)N_NODES * HEADS);
  float* exbuf    = (float*)carve(sizeof(float) * (size_t)EFULL * HEADS);
  float* emb      = (float*)carve(sizeof(float) * (size_t)SEQ * N_NODES * HID);
  float* gibuf    = (float*)carve(sizeof(float) * (size_t)N_NODES * 3 * GRU_H);
  float* ghbuf    = (float*)carve(sizeof(float) * (size_t)N_NODES * 3 * GRU_H);
  float* hf       = (float*)carve(sizeof(float) * (size_t)N_NODES * GRU_H);
  float* hb       = (float*)carve(sizeof(float) * (size_t)N_NODES * GRU_H);
  float* gvec     = (float*)carve(sizeof(float) * 2 * GRU_H);

  auto nblk = [](int n, int t) { return (n + t - 1) / t; };

  auto gemm = [&](const float* A, const float* B, const float* bias, float* Cm,
                  int M, int N, int K, bool transB) {
    dim3 grid(nblk(N, 64), nblk(M, 128));
    if (transB)
      gemm_wmma_f16<1><<<grid, 256, 0, stream>>>(A, B, bias, Cm, M, N, K);
    else
      gemm_wmma_f16<0><<<grid, 256, 0, stream>>>(A, B, bias, Cm, M, N, K);
  };

  // ================= graph setup (self loops + CSR by dst) =================
  zero_u32<<<nblk(N_NODES, 256), 256, 0, stream>>>((unsigned int*)deg, N_NODES);
  zero_f32<<<nblk(N_NODES * 3, 256), 256, 0, stream>>>(loopsum, N_NODES * 3);
  setup_edges<<<nblk(E_EDGES, 256), 256, 0, stream>>>(eattr, esrc, edst, deg, loopsum,
                                                      src_full, dst_full, ea_full, E_EDGES);
  setup_loops<<<nblk(N_NODES, 256), 256, 0, stream>>>(src_full, dst_full, ea_full,
                                                      deg, loopsum, E_EDGES, N_NODES);
  scan_rowptr<<<1, 256, 0, stream>>>(deg, rowptr, N_NODES);
  zero_u32<<<nblk(N_NODES, 256), 256, 0, stream>>>((unsigned int*)fillpos, N_NODES);
  fill_csr<<<nblk(EFULL, 256), 256, 0, stream>>>(dst_full, rowptr, fillpos, eid, EFULL);

  // ================= temporal GAT stack =================
  for (int t = 0; t < SEQ; ++t) {
    const float* xt = x + (size_t)t * N_NODES * IN_CH;
    // ---- GAT layer 1 (128 -> 8x64) ----
    gemm(xt, g1wl, nullptr, xl, N_NODES, HC1, IN_CH, false);
    gemm(xt, g1wr, nullptr, xr, N_NODES, HC1, IN_CH, false);
    zero_u32<<<nblk(N_NODES * HEADS, 256), 256, 0, stream>>>(mx, N_NODES * HEADS);
    zero_f32<<<nblk(N_NODES * HEADS, 256), 256, 0, stream>>>(den, N_NODES * HEADS);
    gat_edge_logits<<<nblk(EFULL, 8), 256, 0, stream>>>(xl, xr, ea_full, src_full, dst_full,
                                                        g1we, g1att, exbuf, mx, EFULL, HEADS, HID);
    gat_edge_softmax<<<nblk(EFULL * HEADS, 256), 256, 0, stream>>>(dst_full, mx, exbuf, den,
                                                                   EFULL, HEADS);
    gat_aggregate<<<N_NODES, HC1, 0, stream>>>(xl, exbuf, den, rowptr, eid, src_full,
                                               g1b, h1, HEADS, HID);
    // ---- GAT layer 2 (512 -> 1x64) ----
    gemm(h1, g2wl, nullptr, xl, N_NODES, HID, HC1, false);
    gemm(h1, g2wr, nullptr, xr, N_NODES, HID, HC1, false);
    zero_u32<<<nblk(N_NODES, 256), 256, 0, stream>>>(mx, N_NODES);
    zero_f32<<<nblk(N_NODES, 256), 256, 0, stream>>>(den, N_NODES);
    gat_edge_logits<<<nblk(EFULL, 8), 256, 0, stream>>>(xl, xr, ea_full, src_full, dst_full,
                                                        g2we, g2att, exbuf, mx, EFULL, 1, HID);
    gat_edge_softmax<<<nblk(EFULL, 256), 256, 0, stream>>>(dst_full, mx, exbuf, den, EFULL, 1);
    gat_aggregate<<<N_NODES, HID, 0, stream>>>(xl, exbuf, den, rowptr, eid, src_full,
                                               g2b, emb + (size_t)t * N_NODES * HID, 1, HID);
  }

  // ================= bidirectional GRU (layer 0 only matters) =================
  auto run_gru = [&](const float* wih, const float* whh, const float* bih,
                     const float* bhh, float* hbuf, bool backward) {
    zero_f32<<<nblk(N_NODES * GRU_H, 256), 256, 0, stream>>>(hbuf, N_NODES * GRU_H);
    for (int s = 0; s < SEQ; ++s) {
      int t = backward ? (SEQ - 1 - s) : s;
      const float* et = emb + (size_t)t * N_NODES * HID;
      gemm(et,   wih, bih, gibuf, N_NODES, 3 * GRU_H, HID,   true);
      gemm(hbuf, whh, bhh, ghbuf, N_NODES, 3 * GRU_H, GRU_H, true);
      gru_cell<<<nblk(N_NODES * GRU_H, 256), 256, 0, stream>>>(gibuf, ghbuf, hbuf,
                                                               N_NODES, GRU_H);
    }
  };
  run_gru(wih_f, whh_f, bih_f, bhh_f, hf, false);
  run_gru(wih_b, whh_b, bih_b, bhh_b, hb, true);

  // ================= node mean + FC =================
  col_mean<<<2 * GRU_H, 256, 0, stream>>>(hf, hb, gvec, N_NODES, GRU_H);
  fc_out<<<1, 64, 0, stream>>>(gvec, fcw, fcb, out);
}
